// GaussiansGenerator_87875030876560
// MI455X (gfx1250) — matrix operand order repaired and checked by reading, not compiled, not run on hardware
//
#include <hip/hip_runtime.h>

// ---------------------------------------------------------------------------
// LINKX x2 for MI455X (gfx1250): edge scatter via fp32 L2 atomics, all dense
// algebra folded into 4 GEMMs/layer executed with v_wmma_f32_16x16x32_bf16.
// LDS staging: A row-major (M x K), B transposed (N x K) so every fragment
// gather is contiguous 32B -> ds_load_b128 pairs.
// ---------------------------------------------------------------------------

#define DD 256      // channels
#define NNODE 16384 // nodes
#define EEDGE 262144
static constexpr float BN_EPS = 1e-5f;

typedef __attribute__((ext_vector_type(16))) __bf16 v16bf;
typedef __attribute__((ext_vector_type(8)))  float  v8f;

// ---------------- scatter: emb = eb + segment_sum(ew[src], dst) -------------

__global__ __launch_bounds__(256) void init_emb_kernel(float* __restrict__ emb,
                                                       const float* __restrict__ eb) {
    int id = blockIdx.x * 256 + threadIdx.x;
    emb[id] = eb[id & (DD - 1)];
}

__global__ __launch_bounds__(256) void scatter_kernel(const int* __restrict__ ei,
                                                      const float* __restrict__ ew,
                                                      float* __restrict__ emb) {
    // 4 edges per block, 64 threads/edge, 4 channels (float4) per thread
    int e   = blockIdx.x * 4 + (threadIdx.x >> 6);
    int t   = threadIdx.x & 63;
    int src = ei[e];
    int dst = ei[EEDGE + e];
    const float4 v = *(const float4*)(ew + (size_t)src * DD + t * 4);
    float* p = emb + (size_t)dst * DD + t * 4;
    atomicAdd(p + 0, v.x);
    atomicAdd(p + 1, v.y);
    atomicAdd(p + 2, v.z);
    atomicAdd(p + 3, v.w);
}

// ---------------- weight fusion preps ---------------------------------------

// W1 = I + c1w ; W2 = nw + nw@c2w ; F1 = f1w   (all -> bf16)
__global__ __launch_bounds__(DD) void prep_w12_kernel(const float* __restrict__ c1w,
                                                      const float* __restrict__ nw,
                                                      const float* __restrict__ c2w,
                                                      const float* __restrict__ f1w,
                                                      __bf16* __restrict__ W1,
                                                      __bf16* __restrict__ W2,
                                                      __bf16* __restrict__ F1) {
    __shared__ float row[DD];
    int i = blockIdx.x, j = threadIdx.x;
    row[j] = nw[i * DD + j];
    __syncthreads();
    float acc = row[j];
    for (int k = 0; k < DD; ++k) acc += row[k] * c2w[k * DD + j];
    W2[i * DD + j] = (__bf16)acc;
    W1[i * DD + j] = (__bf16)(c1w[i * DD + j] + (i == j ? 1.f : 0.f));
    F1[i * DD + j] = (__bf16)f1w[i * DD + j];
}

// b12 = c1b + nb + c2b + nb@c2w
__global__ __launch_bounds__(DD) void prep_b12_kernel(const float* __restrict__ c1b,
                                                      const float* __restrict__ nb,
                                                      const float* __restrict__ c2b,
                                                      const float* __restrict__ c2w,
                                                      float* __restrict__ b12) {
    int j = threadIdx.x;
    float acc = c1b[j] + nb[j] + c2b[j];
    for (int k = 0; k < DD; ++k) acc += nb[k] * c2w[k * DD + j];
    b12[j] = acc;
}

__global__ __launch_bounds__(DD) void zero_stats_kernel(float* a, float* b) {
    a[threadIdx.x] = 0.f;
    b[threadIdx.x] = 0.f;
}

// s = rsqrt(var+eps)*gamma ; betaAdj = beta - mu*s
__global__ __launch_bounds__(DD) void prep_bn_kernel(const float* __restrict__ colsum,
                                                     const float* __restrict__ colsq,
                                                     const float* __restrict__ g,
                                                     const float* __restrict__ b,
                                                     float* __restrict__ sbuf,
                                                     float* __restrict__ betabuf) {
    int j = threadIdx.x;
    float mu  = colsum[j] * (1.f / NNODE);
    float var = colsq[j] * (1.f / NNODE) - mu * mu;
    float s   = rsqrtf(var + BN_EPS) * g[j];
    sbuf[j]    = s;
    betabuf[j] = b[j] - mu * s;
}

// F2s[k][j] = s[k]*f2w[k][j]
__global__ __launch_bounds__(DD) void prep_f2_kernel(const float* __restrict__ f2w,
                                                     const float* __restrict__ sbuf,
                                                     __bf16* __restrict__ F2s) {
    int i = blockIdx.x, j = threadIdx.x;
    F2s[i * DD + j] = (__bf16)(sbuf[i] * f2w[i * DD + j]);
}

// b2 = f2b + betaAdj@f2w
__global__ __launch_bounds__(DD) void prep_b2_kernel(const float* __restrict__ f2w,
                                                     const float* __restrict__ betabuf,
                                                     const float* __restrict__ f2b,
                                                     float* __restrict__ b2f) {
    int j = threadIdx.x;
    float acc = f2b[j];
    for (int k = 0; k < DD; ++k) acc += betabuf[k] * f2w[k * DD + j];
    b2f[j] = acc;
}

// ---------------- WMMA GEMM ------------------------------------------------
// Block: 256 threads = 8 waves (4 Mrows x 2 Ncols); block tile 64(M) x 64(N).
// Each wave owns a 16x32 strip: one shared A fragment feeding two WMMAs.
// K loop in chunks of 32 staged in LDS bf16; A tile M-major, B tile N-major
// (transposed on store) so all fragment gathers are contiguous b128 loads.
// DUAL : C = A1@B1 + A2@B2 (gemm1: emb@W1 + x@W2)
// ABF  : A operand already bf16 in global memory
// RELU : relu after bias
// STATS: accumulate per-column sum/sumsq (BN) via fp32 atomics
// OBF  : store bf16 (intermediates) vs fp32 (with ldout)
template <bool DUAL, bool ABF, bool RELU, bool STATS, bool OBF>
__global__ __launch_bounds__(256) void wmma_gemm_kernel(
    const float* __restrict__ A1f, const __bf16* __restrict__ A1b,
    const float* __restrict__ A2f,
    const __bf16* __restrict__ B1, const __bf16* __restrict__ B2,
    const float* __restrict__ bias,
    float* __restrict__ outF, __bf16* __restrict__ outB, int ldout,
    float* __restrict__ colsum, float* __restrict__ colsq) {

    __shared__ __bf16 sA1[64 * 32];                  // [m][k]
    __shared__ __bf16 sA2[DUAL ? 64 * 32 : 4];
    __shared__ __bf16 sBt1[64 * 32];                 // [n][k] (transposed)
    __shared__ __bf16 sBt2[DUAL ? 64 * 32 : 4];

    const int  tid  = threadIdx.x;
    const int  lane = tid & 31;
    const int  wave = tid >> 5;
    const int  wRow = wave >> 1;   // 0..3
    const int  wCol = wave & 1;    // 0..1
    const bool hi   = lane >= 16;
    const int  ln   = lane & 15;

    const int blockN = blockIdx.x * 64;
    const int blockM = blockIdx.y * 64;

    v8f acc0 = {0.f, 0.f, 0.f, 0.f, 0.f, 0.f, 0.f, 0.f};
    v8f acc1 = {0.f, 0.f, 0.f, 0.f, 0.f, 0.f, 0.f, 0.f};

    const int ao = hi ? 16 : 0;   // A fragment K sub-offset
    const int bo = hi ? 16 : 0;   // B fragment K sub-offset

    for (int kc = 0; kc < DD; kc += 32) {
        // ---- stage A tile(s) 64x32: 4 consecutive K per thread (float4/b64)
#pragma unroll
        for (int r = 0; r < 2; ++r) {
            int c  = tid + r * 256;          // 512 chunks of 4
            int mm = c >> 3, k4 = (c & 7) * 4;
            __bf16* p = &sA1[mm * 32 + k4];
            if constexpr (ABF) {
                const __bf16* s = A1b + (size_t)(blockM + mm) * DD + kc + k4;
                p[0] = s[0]; p[1] = s[1]; p[2] = s[2]; p[3] = s[3];
            } else {
                const float4 v = *(const float4*)(A1f + (size_t)(blockM + mm) * DD + kc + k4);
                p[0] = (__bf16)v.x; p[1] = (__bf16)v.y;
                p[2] = (__bf16)v.z; p[3] = (__bf16)v.w;
            }
            if constexpr (DUAL) {
                const float4 v = *(const float4*)(A2f + (size_t)(blockM + mm) * DD + kc + k4);
                __bf16* q = &sA2[mm * 32 + k4];
                q[0] = (__bf16)v.x; q[1] = (__bf16)v.y;
                q[2] = (__bf16)v.z; q[3] = (__bf16)v.w;
            }
        }
        // ---- stage B tile(s) 32x64 transposed -> [n][k]
#pragma unroll
        for (int r = 0; r < 8; ++r) {
            int idx = tid + r * 256;
            int kk = idx >> 6, nn = idx & 63;
            size_t g = (size_t)(kc + kk) * DD + blockN + nn;
            sBt1[nn * 32 + kk] = B1[g];
            if constexpr (DUAL) sBt2[nn * 32 + kk] = B2[g];
        }
        __syncthreads();

        // Fragments use contiguous K order (K = 16*hi + e), same for A and B. All contiguous:
        //  A: lane row M=ln; e   -> K=ao+e
        //  B: lane col N;    e   -> K=bo+e   (transposed LDS => stride-1)
        v16bf a, b0, b1;
        {
            const __bf16* ar = &sA1[(wRow * 16 + ln) * 32];
#pragma unroll
            for (int e = 0; e < 8; ++e) { a[e] = ar[ao + e]; a[e + 8] = ar[ao + 8 + e]; }
            const __bf16* bc0 = &sBt1[(wCol * 32 + ln) * 32 + bo];
            const __bf16* bc1 = bc0 + 16 * 32;
#pragma unroll
            for (int e = 0; e < 16; ++e) { b0[e] = bc0[e]; b1[e] = bc1[e]; }
        }
        acc0 = __builtin_amdgcn_wmma_f32_16x16x32_bf16(false, a, false, b0,
                                                       (short)0, acc0, false, false);
        acc1 = __builtin_amdgcn_wmma_f32_16x16x32_bf16(false, a, false, b1,
                                                       (short)0, acc1, false, false);
        if constexpr (DUAL) {
            const __bf16* ar = &sA2[(wRow * 16 + ln) * 32];
#pragma unroll
            for (int e = 0; e < 8; ++e) { a[e] = ar[ao + e]; a[e + 8] = ar[ao + 8 + e]; }
            const __bf16* bc0 = &sBt2[(wCol * 32 + ln) * 32 + bo];
            const __bf16* bc1 = bc0 + 16 * 32;
#pragma unroll
            for (int e = 0; e < 16; ++e) { b0[e] = bc0[e]; b1[e] = bc1[e]; }
            acc0 = __builtin_amdgcn_wmma_f32_16x16x32_bf16(false, a, false, b0,
                                                           (short)0, acc0, false, false);
            acc1 = __builtin_amdgcn_wmma_f32_16x16x32_bf16(false, a, false, b1,
                                                           (short)0, acc1, false, false);
        }
        __syncthreads();
    }

    // C/D layout: lanes 0-15 -> M=v, lanes 16-31 -> M=v+8; N = lane&15
    const int rowBase = blockM + wRow * 16 + (hi ? 8 : 0);
#pragma unroll
    for (int t = 0; t < 2; ++t) {
        const v8f&  av  = t ? acc1 : acc0;
        const int   col = blockN + wCol * 32 + t * 16 + ln;
        const float bv  = bias[col];
        float ps = 0.f, psq = 0.f;
#pragma unroll
        for (int v = 0; v < 8; ++v) {
            int   row = rowBase + v;
            float val = av[v] + bv;
            if constexpr (RELU) val = fmaxf(val, 0.f);
            if constexpr (STATS) { ps += val; psq += val * val; }
            if constexpr (OBF) outB[(size_t)row * DD + col] = (__bf16)val;
            else               outF[(size_t)row * ldout + col] = val;
        }
        if constexpr (STATS) {
            atomicAdd(&colsum[col], ps);
            atomicAdd(&colsq[col], psq);
        }
    }
}

// ---------------- concat copy: out[:, D:2D] = x -----------------------------

__global__ __launch_bounds__(256) void copy_x_kernel(const float* __restrict__ x,
                                                     float* __restrict__ out) {
    int id = blockIdx.x * 256 + threadIdx.x;
    int row = id >> 8, col = id & 255;
    out[(size_t)row * (2 * DD) + DD + col] = x[id];
}

// ---------------------------------------------------------------------------

extern "C" void kernel_launch(void* const* d_in, const int* in_sizes, int n_in,
                              void* d_out, int out_size, void* d_ws, size_t ws_size,
                              hipStream_t stream) {
    (void)in_sizes; (void)n_in; (void)out_size; (void)ws_size;

    const float* x  = (const float*)d_in[0];
    const int*   ei = (const int*)d_in[2];

    // workspace layout
    char* w = (char*)d_ws;
    auto alloc = [&](size_t bytes) -> void* {
        void* p = (void*)w;
        w += (bytes + 255) & ~(size_t)255;
        return p;
    };
    float*  emb  = (float*)alloc((size_t)NNODE * DD * 4);
    float*  x1   = (float*)alloc((size_t)NNODE * DD * 4); // layer-1 output
    __bf16* T1   = (__bf16*)alloc((size_t)NNODE * DD * 2);
    __bf16* W1   = (__bf16*)alloc((size_t)DD * DD * 2);
    __bf16* W2   = (__bf16*)alloc((size_t)DD * DD * 2);
    __bf16* F1   = (__bf16*)alloc((size_t)DD * DD * 2);
    __bf16* F2s  = (__bf16*)alloc((size_t)DD * DD * 2);
    float*  b12  = (float*)alloc(DD * 4);
    float*  b2f  = (float*)alloc(DD * 4);
    float*  csum = (float*)alloc(DD * 4);
    float*  csq  = (float*)alloc(DD * 4);
    float*  sbuf = (float*)alloc(DD * 4);
    float*  bbuf = (float*)alloc(DD * 4);
    __bf16* Hb   = (__bf16*)emb; // emb is dead after gemm1; reuse its 16MB for H

    const dim3 gGemm(DD / 64, NNODE / 64);

    auto layer = [&](const float* xin, int pb, float* outp, int ldout) {
        const float* ew  = (const float*)d_in[pb + 0];
        const float* eb  = (const float*)d_in[pb + 1];
        const float* nw  = (const float*)d_in[pb + 2];
        const float* nb  = (const float*)d_in[pb + 3];
        const float* c1w = (const float*)d_in[pb + 4];
        const float* c1b = (const float*)d_in[pb + 5];
        const float* c2w = (const float*)d_in[pb + 6];
        const float* c2b = (const float*)d_in[pb + 7];
        const float* f1w = (const float*)d_in[pb + 8];
        const float* f1b = (const float*)d_in[pb + 9];
        const float* bng = (const float*)d_in[pb + 10];
        const float* bnb = (const float*)d_in[pb + 11];
        const float* f2w = (const float*)d_in[pb + 12];
        const float* f2b = (const float*)d_in[pb + 13];

        // emb = eb + segment_sum(ew[src], dst)
        init_emb_kernel<<<NNODE * DD / 256, 256, 0, stream>>>(emb, eb);
        scatter_kernel<<<EEDGE / 4, 256, 0, stream>>>(ei, ew, emb);

        // fused weights: W1 = I + c1w, W2 = nw + nw@c2w, F1 = f1w
        prep_w12_kernel<<<DD, DD, 0, stream>>>(c1w, nw, c2w, f1w, W1, W2, F1);
        prep_b12_kernel<<<1, DD, 0, stream>>>(c1b, nb, c2b, c2w, b12);

        // T1 = relu(emb@W1 + xin@W2 + b12)   (bf16)
        wmma_gemm_kernel<true, false, true, false, true>
            <<<gGemm, 256, 0, stream>>>(emb, nullptr, xin, W1, W2, b12,
                                        nullptr, T1, DD, nullptr, nullptr);

        // H = relu(T1@F1 + f1b), accumulate BN column stats
        zero_stats_kernel<<<1, DD, 0, stream>>>(csum, csq);
        wmma_gemm_kernel<false, true, true, true, true>
            <<<gGemm, 256, 0, stream>>>(nullptr, T1, nullptr, F1, nullptr, f1b,
                                        nullptr, Hb, DD, csum, csq);

        // fold BN affine into F2: F2s = diag(s)@f2w, b2 = f2b + betaAdj@f2w
        prep_bn_kernel<<<1, DD, 0, stream>>>(csum, csq, bng, bnb, sbuf, bbuf);
        prep_f2_kernel<<<DD, DD, 0, stream>>>(f2w, sbuf, F2s);
        prep_b2_kernel<<<1, DD, 0, stream>>>(f2w, bbuf, f2b, b2f);

        // out = H@F2s + b2   (fp32, strided by ldout)
        wmma_gemm_kernel<false, true, false, false, false>
            <<<gGemm, 256, 0, stream>>>(nullptr, Hb, nullptr, F2s, nullptr, b2f,
                                        outp, nullptr, ldout, nullptr, nullptr);
    };

    layer(x, 4, x1, DD);
    layer(x1, 18, (float*)d_out, 2 * DD);
    copy_x_kernel<<<NNODE * DD / 256, 256, 0, stream>>>(x, (float*)d_out);
}
